// HybridFeedForward_45964740002183
// MI455X (gfx1250) — matrix operand order repaired and checked by reading, not compiled, not run on hardware
//
#include <hip/hip_runtime.h>
#include <hip/hip_bf16.h>
#include <math.h>

typedef __attribute__((ext_vector_type(16))) __bf16 v16bf;
typedef __attribute__((ext_vector_type(8)))  float  v8f;

#define TPB 256
#define BK 32
#define BKP 36   // padded LDS K pitch (72 bytes/row: 8B-aligned, conflict-light)

// ---------------------------------------------------------------------------
// TT weight reconstruction (3 tiny stages, R1=12, R2=24, R3=12 fixed)
//   M[i1,o1,i2,o2,r2] = sum_r1 ga[i1,o1,r1] * gb[r1,i2,o2,r2]
//   V[r2,i3,o3,i4,o4] = sum_r3 gc[r2,i3,o3,r3] * gd[r3,i4,o4]
//   Wt[o,i]           = sum_r2 M[...] * V[...]   (stored transposed, K-contig,
//                                                 split into bf16 hi/lo pair)
// ---------------------------------------------------------------------------
__global__ __launch_bounds__(TPB) void tt_r1(const float* __restrict__ ga,
                                             const float* __restrict__ gb,
                                             float* __restrict__ Mt,
                                             int I1, int O1, int I2, int O2) {
  int tid = blockIdx.x * blockDim.x + threadIdx.x;
  int total = I1 * O1 * I2 * O2 * 24;
  if (tid >= total) return;
  int r2 = tid % 24; int t = tid / 24;
  int o2 = t % O2; t /= O2;
  int i2 = t % I2; t /= I2;
  int o1 = t % O1; int i1 = t / O1;
  float s = 0.f;
#pragma unroll
  for (int r1 = 0; r1 < 12; ++r1)
    s += ga[(i1 * O1 + o1) * 12 + r1] * gb[((r1 * I2 + i2) * O2 + o2) * 24 + r2];
  Mt[tid] = s;
}

__global__ __launch_bounds__(TPB) void tt_r2(const float* __restrict__ gc,
                                             const float* __restrict__ gd,
                                             float* __restrict__ Vt,
                                             int I3, int O3, int I4, int O4) {
  int tid = blockIdx.x * blockDim.x + threadIdx.x;
  int total = 24 * I3 * O3 * I4 * O4;
  if (tid >= total) return;
  int o4 = tid % O4; int t = tid / O4;
  int i4 = t % I4; t /= I4;
  int o3 = t % O3; t /= O3;
  int i3 = t % I3; int r2 = t / I3;
  float s = 0.f;
#pragma unroll
  for (int r3 = 0; r3 < 12; ++r3)
    s += gc[((r2 * I3 + i3) * O3 + o3) * 12 + r3] * gd[(r3 * I4 + i4) * O4 + o4];
  Vt[tid] = s;
}

__global__ __launch_bounds__(TPB) void tt_r3(const float* __restrict__ Mt,
                                             const float* __restrict__ Vt,
                                             __bf16* __restrict__ Whi,
                                             __bf16* __restrict__ Wlo,
                                             int I1, int I2, int I3, int I4,
                                             int O1, int O2, int O3, int O4,
                                             int K, int Nout) {
  int tid = blockIdx.x * blockDim.x + threadIdx.x;
  if (tid >= K * Nout) return;
  int i = tid % K;   // K contiguous -> coalesced Wt writes
  int o = tid / K;
  int i4 = i % I4; int t = i / I4;
  int i3 = t % I3; t /= I3;
  int i2 = t % I2; int i1 = t / I2;
  int o4 = o % O4; int u = o / O4;
  int o3 = u % O3; u /= O3;
  int o2 = u % O2; int o1 = u / O2;
  const float* mp = Mt + (((i1 * O1 + o1) * I2 + i2) * O2 + o2) * 24;
  int vbase   = ((i3 * O3 + o3) * I4 + i4) * O4 + o4;
  int vstride = I3 * O3 * I4 * O4;
  float s = 0.f;
#pragma unroll
  for (int r2 = 0; r2 < 24; ++r2) s += mp[r2] * Vt[vbase + r2 * vstride];
  __bf16 hi = (__bf16)s;
  __bf16 lo = (__bf16)(s - (float)hi);
  Whi[tid] = hi;
  Wlo[tid] = lo;
}

// ---------------------------------------------------------------------------
// Split-precision GEMM:  C = A(f32, MxK) * W(bf16 hi/lo, transposed NxK) + bias
// bf16x3: acc += Ahi*Bhi + Alo*Bhi + Ahi*Blo   (f32 accumulate, ~fp32 accuracy)
// Double-buffered LDS with register prefetch: one barrier per K-step; global
// loads for step k+1 issue before the 24 (or 12) WMMAs of step k.
//   WM_ x WN_ waves (8 total), each owning MT_ x NT_ 16x16 fragments.
// ---------------------------------------------------------------------------
union BfPack4 { __bf16 h[4]; uint2 u; };
union FragU   { unsigned int u[8]; v16bf v; };

template <bool GELU, int WM_, int WN_, int MT_, int NT_>
__global__ __launch_bounds__(TPB) void tt_gemm(const float*  __restrict__ A,
                                               const __bf16* __restrict__ Bhi,
                                               const __bf16* __restrict__ Blo,
                                               const float*  __restrict__ bias,
                                               float* __restrict__ C,
                                               int M, int N, int K) {
  constexpr int BM  = WM_ * MT_ * 16;
  constexpr int BN  = WN_ * NT_ * 16;
  constexpr int APT = BM / 32;   // float4 loads per thread (A tile)
  constexpr int BPT = BN / 32;   // uint2 loads per thread per array (B tile)

  __shared__ __attribute__((aligned(16))) __bf16 sAhi[2][BM][BKP];
  __shared__ __attribute__((aligned(16))) __bf16 sAlo[2][BM][BKP];
  __shared__ __attribute__((aligned(16))) __bf16 sBhi[2][BN][BKP];
  __shared__ __attribute__((aligned(16))) __bf16 sBlo[2][BN][BKP];

  const int tid   = threadIdx.x;
  const int lane  = tid & 31;
  const int wave  = tid >> 5;          // 0..7
  const int wm    = wave % WM_;
  const int wn    = wave / WM_;
  const int lrow  = lane & 15;
  const int lhalf = lane >> 4;
  const int tileM = blockIdx.y * BM;
  const int tileN = blockIdx.x * BN;

  // per-thread staging coordinates (hoisted out of the K loop)
  int aRow[APT], aKv[APT];
#pragma unroll
  for (int i = 0; i < APT; ++i) {
    int v = tid + i * TPB;
    aRow[i] = v >> 3;
    aKv[i]  = v & 7;
  }
  int bRow[BPT], bKv[BPT];
#pragma unroll
  for (int i = 0; i < BPT; ++i) {
    int v = tid + i * TPB;
    bRow[i] = v >> 3;
    bKv[i]  = v & 7;
  }

  v8f acc[MT_][NT_];
#pragma unroll
  for (int mt = 0; mt < MT_; ++mt)
#pragma unroll
    for (int nt = 0; nt < NT_; ++nt) acc[mt][nt] = (v8f)0.f;

  float4 aReg[APT];
  uint2  bhReg[BPT], blReg[BPT];

  // ---- prologue: fetch tile k=0 ----
#pragma unroll
  for (int i = 0; i < APT; ++i)
    aReg[i] = *(const float4*)(A + (size_t)(tileM + aRow[i]) * K + aKv[i] * 4);
#pragma unroll
  for (int i = 0; i < BPT; ++i) {
    size_t g = ((size_t)(tileN + bRow[i]) * K) / 4 + bKv[i];
    bhReg[i] = ((const uint2*)Bhi)[g];
    blReg[i] = ((const uint2*)Blo)[g];
  }
  // store tile 0 into buffer 0 (fp32 -> hi/lo bf16 for A)
#pragma unroll
  for (int i = 0; i < APT; ++i) {
    float fs[4] = {aReg[i].x, aReg[i].y, aReg[i].z, aReg[i].w};
    BfPack4 ph, pl;
#pragma unroll
    for (int j = 0; j < 4; ++j) {
      __bf16 h = (__bf16)fs[j];
      ph.h[j] = h;
      pl.h[j] = (__bf16)(fs[j] - (float)h);
    }
    *(uint2*)&sAhi[0][aRow[i]][aKv[i] * 4] = ph.u;
    *(uint2*)&sAlo[0][aRow[i]][aKv[i] * 4] = pl.u;
  }
#pragma unroll
  for (int i = 0; i < BPT; ++i) {
    *(uint2*)&sBhi[0][bRow[i]][bKv[i] * 4] = bhReg[i];
    *(uint2*)&sBlo[0][bRow[i]][bKv[i] * 4] = blReg[i];
  }
  __syncthreads();

  int p = 0;
  for (int k0 = 0; k0 < K; k0 += BK) {
    const int kn = k0 + BK;
    const bool more = kn < K;

    // ---- prefetch next tile into registers (overlaps with WMMAs below) ----
    if (more) {
#pragma unroll
      for (int i = 0; i < APT; ++i)
        aReg[i] = *(const float4*)(A + (size_t)(tileM + aRow[i]) * K + kn + aKv[i] * 4);
#pragma unroll
      for (int i = 0; i < BPT; ++i) {
        size_t g = ((size_t)(tileN + bRow[i]) * K + kn) / 4 + bKv[i];
        bhReg[i] = ((const uint2*)Bhi)[g];
        blReg[i] = ((const uint2*)Blo)[g];
      }
    }

    // ---- load fragments from LDS buffer p (ISA 7.12.2 layouts) ----
    v16bf afh[MT_], afl[MT_], bfh[NT_], bfl[NT_];
#pragma unroll
    for (int mt = 0; mt < MT_; ++mt) {
      int r  = wm * (MT_ * 16) + mt * 16 + lrow;  // A row = lane&15 both halves
      int kb = lhalf * 8;                         // lanes16-31: K=8..15 / 24..31
      FragU fh, fl;
#pragma unroll
      for (int j = 0; j < 4; ++j) {
        fh.u[j]     = *(const unsigned int*)&sAhi[p][r][kb + 2 * j];
        fh.u[4 + j] = *(const unsigned int*)&sAhi[p][r][16 + kb + 2 * j];
        fl.u[j]     = *(const unsigned int*)&sAlo[p][r][kb + 2 * j];
        fl.u[4 + j] = *(const unsigned int*)&sAlo[p][r][16 + kb + 2 * j];
      }
      afh[mt] = fh.v;
      afl[mt] = fl.v;
    }
#pragma unroll
    for (int nt = 0; nt < NT_; ++nt) {
      int c  = wn * (NT_ * 16) + nt * 16 + lrow;  // B col = lane&15
      int ks = lhalf * 8;                         // lanes16-31: K=8..15 / 24..31
      FragU gh, gl;
#pragma unroll
      for (int j = 0; j < 4; ++j) {
        gh.u[j]     = *(const unsigned int*)&sBhi[p][c][ks + 2 * j];
        gh.u[4 + j] = *(const unsigned int*)&sBhi[p][c][16 + ks + 2 * j];
        gl.u[j]     = *(const unsigned int*)&sBlo[p][c][ks + 2 * j];
        gl.u[4 + j] = *(const unsigned int*)&sBlo[p][c][16 + ks + 2 * j];
      }
      bfh[nt] = gh.v;
      bfl[nt] = gl.v;
    }

    // ---- bf16x3 split-precision WMMAs ----
#pragma unroll
    for (int mt = 0; mt < MT_; ++mt)
#pragma unroll
      for (int nt = 0; nt < NT_; ++nt) {
        acc[mt][nt] = __builtin_amdgcn_wmma_f32_16x16x32_bf16(
            false, afh[mt], false, bfh[nt], (short)0, acc[mt][nt], false, false);
        acc[mt][nt] = __builtin_amdgcn_wmma_f32_16x16x32_bf16(
            false, afl[mt], false, bfh[nt], (short)0, acc[mt][nt], false, false);
        acc[mt][nt] = __builtin_amdgcn_wmma_f32_16x16x32_bf16(
            false, afh[mt], false, bfl[nt], (short)0, acc[mt][nt], false, false);
      }

    // ---- drain prefetch into the other buffer; single barrier per step ----
    if (more) {
      const int q = p ^ 1;
#pragma unroll
      for (int i = 0; i < APT; ++i) {
        float fs[4] = {aReg[i].x, aReg[i].y, aReg[i].z, aReg[i].w};
        BfPack4 ph, pl;
#pragma unroll
        for (int j = 0; j < 4; ++j) {
          __bf16 h = (__bf16)fs[j];
          ph.h[j] = h;
          pl.h[j] = (__bf16)(fs[j] - (float)h);
        }
        *(uint2*)&sAhi[q][aRow[i]][aKv[i] * 4] = ph.u;
        *(uint2*)&sAlo[q][aRow[i]][aKv[i] * 4] = pl.u;
      }
#pragma unroll
      for (int i = 0; i < BPT; ++i) {
        *(uint2*)&sBhi[q][bRow[i]][bKv[i] * 4] = bhReg[i];
        *(uint2*)&sBlo[q][bRow[i]][bKv[i] * 4] = blReg[i];
      }
      __syncthreads();
      p = q;
    }
  }

  // ---- epilogue: bias (+ exact-erf GELU), fp32 store ----
#pragma unroll
  for (int mt = 0; mt < MT_; ++mt)
#pragma unroll
    for (int nt = 0; nt < NT_; ++nt) {
      int col = tileN + wn * (NT_ * 16) + nt * 16 + lrow;
      float bv = bias[col];
#pragma unroll
      for (int j = 0; j < 8; ++j) {
        int row = tileM + wm * (MT_ * 16) + mt * 16 + lhalf * 8 + j;
        float v = acc[mt][nt][j] + bv;
        if (GELU) v = 0.5f * v * (1.0f + erff(v * 0.70710678118654752f));
        C[(size_t)row * N + col] = v;
      }
    }
}

// ---------------------------------------------------------------------------
extern "C" void kernel_launch(void* const* d_in, const int* in_sizes, int n_in,
                              void* d_out, int out_size, void* d_ws, size_t ws_size,
                              hipStream_t stream) {
  const float* x   = (const float*)d_in[0];
  const float* g1a = (const float*)d_in[1];
  const float* g1b = (const float*)d_in[2];
  const float* g1c = (const float*)d_in[3];
  const float* g1d = (const float*)d_in[4];
  const float* b1  = (const float*)d_in[5];
  const float* g2a = (const float*)d_in[6];
  const float* g2b = (const float*)d_in[7];
  const float* g2c = (const float*)d_in[8];
  const float* g2d = (const float*)d_in[9];
  const float* b2  = (const float*)d_in[10];

  const int NTOK = 2 * 2048;          // 4096 rows
  const int D1 = 768, D2 = 3072;

  char* ws = (char*)d_ws;
  size_t off = 0;
  float*  H    = (float*)(ws + off);  off += (size_t)NTOK * D2 * 4;   // 50.3 MB
  __bf16* W1hi = (__bf16*)(ws + off); off += (size_t)D1 * D2 * 2;
  __bf16* W1lo = (__bf16*)(ws + off); off += (size_t)D1 * D2 * 2;
  __bf16* W2hi = (__bf16*)(ws + off); off += (size_t)D1 * D2 * 2;
  __bf16* W2lo = (__bf16*)(ws + off); off += (size_t)D1 * D2 * 2;
  float*  Mt   = (float*)(ws + off);  off += 9216 * 4;
  float*  Vt   = (float*)(ws + off);  off += 147456 * 4;

  // --- reconstruct W1 (768x3072), in_f=(4,4,6,8), out_f=(4,6,8,16) ---
  tt_r1<<<(9216 + TPB - 1) / TPB, TPB, 0, stream>>>(g1a, g1b, Mt, 4, 4, 4, 6);
  tt_r2<<<(147456 + TPB - 1) / TPB, TPB, 0, stream>>>(g1c, g1d, Vt, 6, 8, 8, 16);
  tt_r3<<<(D1 * D2 + TPB - 1) / TPB, TPB, 0, stream>>>(Mt, Vt, W1hi, W1lo,
                                                       4, 4, 6, 8, 4, 6, 8, 16,
                                                       D1, D2);
  // --- reconstruct W2 (3072x768), in_f=(4,6,8,16), out_f=(4,4,6,8) ---
  tt_r1<<<(9216 + TPB - 1) / TPB, TPB, 0, stream>>>(g2a, g2b, Mt, 4, 4, 6, 4);
  tt_r2<<<(147456 + TPB - 1) / TPB, TPB, 0, stream>>>(g2c, g2d, Vt, 8, 6, 16, 8);
  tt_r3<<<(D1 * D2 + TPB - 1) / TPB, TPB, 0, stream>>>(Mt, Vt, W2hi, W2lo,
                                                       4, 6, 8, 16, 4, 4, 6, 8,
                                                       D2, D1);

  // --- GEMM1: H = gelu(x @ W1 + b1)   (4096x768 @ 768x3072), 128x128 tiles ---
  dim3 grid1(D2 / 128, NTOK / 128);
  tt_gemm<true, 4, 2, 2, 4><<<grid1, TPB, 0, stream>>>(x, W1hi, W1lo, b1, H,
                                                       NTOK, D2, D1);

  // --- GEMM2: y = H @ W2 + b2   (4096x3072 @ 3072x768), 64x128 tiles ---
  dim3 grid2(D1 / 128, NTOK / 64);
  tt_gemm<false, 2, 4, 2, 2><<<grid2, TPB, 0, stream>>>(H, W2hi, W2lo, b2,
                                                        (float*)d_out,
                                                        NTOK, D1, D2);
}
